// PIAA_ICI_65609920413986
// MI455X (gfx1250) — matrix operand order repaired and checked by reading, not compiled, not run on hardware
//
#include <hip/hip_runtime.h>
#include <hip/hip_bf16.h>
#include <math.h>

// ---------------------------------------------------------------------------
// Types for WMMA fragments (CDNA5 / gfx1250, wave32)
// ---------------------------------------------------------------------------
typedef _Float16 v8h  __attribute__((ext_vector_type(8)));
typedef _Float16 v16h __attribute__((ext_vector_type(16)));
typedef float    v8f  __attribute__((ext_vector_type(8)));

#define BM 64      // block tile rows (output pixels)
#define BNN 64     // block tile cols (output channels)
#define BK 64      // K staged per iteration (2 wmma k-substeps of 32)
#define APITCH 72  // halves; 144B row pitch -> 16B aligned sub-reads
#define BPITCH 72  // halves; 144B row pitch -> 16B aligned sub-reads

static __device__ __forceinline__ v16h cat16(v8h lo, v8h hi) {
  return __builtin_shufflevector(lo, hi, 0,1,2,3,4,5,6,7,8,9,10,11,12,13,14,15);
}

// ---------------------------------------------------------------------------
// Implicit-GEMM convolution with fused BN affine + residual + ReLU.
//   X : f16 activations, NCHW
//   Wt: f16 weights, [Co][K] with K = Ci*R*S (OIHW flatten == this order)
//   Y : f16 output, NCHW
// GEMM: C[M=N*Ho*Wo, Co] = A[M,K] * B[K,Co], f32 accumulate via WMMA.
// 8 waves; wave (wm,wn) owns a 16x32 C tile -> 4 v_wmma per K-stage.
// B tile is streamed into LDS with GLOBAL_LOAD_ASYNC_TO_LDS_B128 (ASYNCcnt).
// ---------------------------------------------------------------------------
__global__ __launch_bounds__(256)
void k_conv_wmma(const _Float16* __restrict__ X, const _Float16* __restrict__ Wt,
                 const float* __restrict__ bnScale, const float* __restrict__ bnBias,
                 const _Float16* __restrict__ resid, _Float16* __restrict__ Y,
                 int Nb, int Ci, int Hi, int Wi, int Co, int Ho, int Wo,
                 int R, int S, int stride, int pad, int relu)
{
  __shared__ __align__(16) _Float16 At[BM * APITCH];
  __shared__ __align__(16) _Float16 Bt[BNN * BPITCH];

  const int K  = Ci * R * S;
  const int RS = R * S;
  const int M  = Nb * Ho * Wo;
  const int HW = Ho * Wo;
  const int HiWi = Hi * Wi;

  const int mBase = blockIdx.x * BM;
  const int nBase = blockIdx.y * BNN;

  const int t    = threadIdx.x;
  const int lane = t & 31;
  const int wv   = t >> 5;        // 8 waves
  const int wm   = wv & 3;        // 4 wave-rows of 16 (M)
  const int wn   = wv >> 2;       // 2 wave-cols of 32 (N)

  // A-tile gather assignment: 4 threads per row, 16 consecutive k each
  const int arow = t >> 2;
  const int koff = (t & 3) * 16;
  const int m = mBase + arow;
  const bool mvalid = (m < M);
  int n0 = 0, oy = 0, ox = 0;
  if (mvalid) { n0 = m / HW; int r = m - n0 * HW; oy = r / Wo; ox = r - oy * Wo; }
  const int iy0 = oy * stride - pad;
  const int ix0 = ox * stride - pad;
  const _Float16* __restrict__ Xn = X + (size_t)n0 * Ci * HiWi;

  // B-tile async-copy assignment: 512 chunks of 16B; thread t moves chunks 2t, 2t+1
  const uint32_t ldsB = (uint32_t)(size_t)(void*)Bt;   // LDS byte offset (addr[31:0])
  const int c0 = t * 2;
  const int brow0 = c0 >> 3;          // 8 chunks (128B) per co row
  const int boff0 = (c0 & 7) * 16;    // byte offset within row
  const int brow1 = (c0 + 1) >> 3;
  const int boff1 = ((c0 + 1) & 7) * 16;

  v8f acc0 = {};
  v8f acc1 = {};

  for (int kb = 0; kb < K; kb += BK) {
    const bool full = (kb + BK <= K);
    // ---- stage B tile ([co][k] halves in LDS) ----
    if (full) {
      uint32_t l0 = ldsB + (uint32_t)(brow0 * (BPITCH * 2) + boff0);
      uint32_t g0 = ((uint32_t)(nBase + brow0) * (uint32_t)K + (uint32_t)kb) * 2u + (uint32_t)boff0;
      uint32_t l1 = ldsB + (uint32_t)(brow1 * (BPITCH * 2) + boff1);
      uint32_t g1 = ((uint32_t)(nBase + brow1) * (uint32_t)K + (uint32_t)kb) * 2u + (uint32_t)boff1;
      asm volatile("global_load_async_to_lds_b128 %0, %1, %2 offset:0"
                   :: "v"(l0), "v"(g0), "s"(Wt) : "memory");
      asm volatile("global_load_async_to_lds_b128 %0, %1, %2 offset:0"
                   :: "v"(l1), "v"(g1), "s"(Wt) : "memory");
    } else {
      // ragged-K tail (stem only): zero-padded scalar path
      const int row = t >> 2;
      const _Float16* wrow = Wt + (size_t)(nBase + row) * K;
      __builtin_prefetch(wrow + kb, 0, 0);
#pragma unroll
      for (int u = 0; u < 16; ++u) {
        int k = kb + koff + u;
        Bt[row * BPITCH + koff + u] = (k < K) ? wrow[k] : (_Float16)0.0f;
      }
    }

    // ---- stage A tile (im2col gather, vectorized LDS store) ----
#pragma unroll
    for (int half = 0; half < 2; ++half) {
      v8h vals = {};
      const int kb0 = kb + koff + half * 8;
      if (mvalid) {
        if (RS == 1) {                    // 1x1 conv fast path: k == ci
          const _Float16* p = Xn + (size_t)kb0 * HiWi + (size_t)iy0 * Wi + ix0;
#pragma unroll
          for (int u = 0; u < 8; ++u)
            if (kb0 + u < K) vals[u] = p[(size_t)u * HiWi];
        } else {
#pragma unroll
          for (int u = 0; u < 8; ++u) {
            int k = kb0 + u;
            if (k < K) {
              int ci = k / RS;
              int rs = k - ci * RS;
              int ry = rs / S;
              int sx = rs - ry * S;
              int iy = iy0 + ry, ix = ix0 + sx;
              if ((unsigned)iy < (unsigned)Hi && (unsigned)ix < (unsigned)Wi)
                vals[u] = Xn[(size_t)ci * HiWi + (size_t)iy * Wi + ix];
            }
          }
        }
      }
      *(v8h*)(&At[arow * APITCH + koff + half * 8]) = vals;
    }

    if (full) asm volatile("s_wait_asynccnt 0x0" ::: "memory");
    __syncthreads();

    // ---- fragments per ISA 7.12.2 layouts (wave32), 2 K-substeps ----
    const int khalf = lane >> 4;
    const int l15 = lane & 15;
    const _Float16* ap  = &At[(wm * 16 + l15) * APITCH];
    const _Float16* bp0 = &Bt[(wn * 32 + l15) * BPITCH];        // N-subtile 0
    const _Float16* bp1 = &Bt[(wn * 32 + 16 + l15) * BPITCH];   // N-subtile 1
#pragma unroll
    for (int ks = 0; ks < 2; ++ks) {
      const int kof = ks * 32;
      v8h a_lo = *(const v8h*)(ap + kof + khalf * 8);        // e 0-7 : K=khalf*8+e
      v8h a_hi = *(const v8h*)(ap + kof + 16 + khalf * 8);   // e 8-15: K=16+khalf*8+e
      v16h af = cat16(a_lo, a_hi);

      v8h b0lo = *(const v8h*)(bp0 + kof + khalf * 8);       // e 0-7: K=khalf*8+e; e 8-15: K=16+khalf*8+e
      v8h b0hi = *(const v8h*)(bp0 + kof + 16 + khalf * 8);
      acc0 = __builtin_amdgcn_wmma_f32_16x16x32_f16(
          false, af, false, cat16(b0lo, b0hi), (short)0, acc0, false, false);

      v8h b1lo = *(const v8h*)(bp1 + kof + khalf * 8);
      v8h b1hi = *(const v8h*)(bp1 + kof + 16 + khalf * 8);
      acc1 = __builtin_amdgcn_wmma_f32_16x16x32_f16(
          false, af, false, cat16(b1lo, b1hi), (short)0, acc1, false, false);
    }
    __syncthreads();
  }

  // ---- epilogue: BN affine + residual + ReLU, store f16 NCHW ----
  const int l15 = lane & 15;
#pragma unroll
  for (int ns = 0; ns < 2; ++ns) {
    const v8f acc = ns ? acc1 : acc0;
    const int co = nBase + wn * 32 + ns * 16 + l15;
    const float sc = bnScale ? bnScale[co] : 1.0f;
    const float bi = bnBias ? bnBias[co] : 0.0f;
#pragma unroll
    for (int v = 0; v < 8; ++v) {
      int mr = mBase + wm * 16 + (lane >> 4) * 8 + v;  // C layout: M = v + lanehalf*8
      if (mr >= M) continue;
      int n2 = mr / HW; int r2 = mr - n2 * HW;
      int oy2 = r2 / Wo, ox2 = r2 - oy2 * Wo;
      size_t oidx = ((size_t)n2 * Co + co) * HW + (size_t)oy2 * Wo + ox2;
      float val = acc[v] * sc + bi;
      if (resid) val += (float)resid[oidx];
      if (relu && val < 0.0f) val = 0.0f;
      Y[oidx] = (_Float16)val;
    }
  }
}

// ---------------------------------------------------------------------------
// Small utility kernels
// ---------------------------------------------------------------------------
__global__ void k_cvt_f16(const float* __restrict__ x, _Float16* __restrict__ y, size_t n) {
  size_t i = (size_t)blockIdx.x * blockDim.x + threadIdx.x;
  if (i < n) y[i] = (_Float16)x[i];
}

__global__ void k_fill0(float* __restrict__ y, int n) {
  int i = blockIdx.x * blockDim.x + threadIdx.x;
  if (i < n) y[i] = 0.0f;
}

__global__ void k_maxpool3s2(const _Float16* __restrict__ X, _Float16* __restrict__ Y,
                             int NC, int Hi, int Wi, int Ho, int Wo) {
  int i = blockIdx.x * blockDim.x + threadIdx.x;
  int total = NC * Ho * Wo;
  if (i >= total) return;
  int ox = i % Wo; int tt = i / Wo; int oy = tt % Ho; int nc = tt / Ho;
  const _Float16* Xp = X + (size_t)nc * Hi * Wi;
  float mx = -3.4e38f;
  for (int ry = 0; ry < 3; ++ry) {
    int iy = oy * 2 - 1 + ry;
    if ((unsigned)iy >= (unsigned)Hi) continue;
    for (int sx = 0; sx < 3; ++sx) {
      int ix = ox * 2 - 1 + sx;
      if ((unsigned)ix >= (unsigned)Wi) continue;
      float v = (float)Xp[(size_t)iy * Wi + ix];
      mx = v > mx ? v : mx;
    }
  }
  Y[i] = (_Float16)mx;
}

__global__ void k_avgpool(const _Float16* __restrict__ X, float* __restrict__ Y,
                          int NC, int HW) {
  int i = blockIdx.x * blockDim.x + threadIdx.x;
  if (i >= NC) return;
  const _Float16* Xp = X + (size_t)i * HW;
  float s = 0.0f;
  for (int j = 0; j < HW; ++j) s += (float)Xp[j];
  Y[i] = s / (float)HW;
}

// Y[m][o] = sum_i X[m][i] * (trans ? W[o][i] : W[i][o]) + B[o]   (+ReLU)
__global__ void k_linear(const float* __restrict__ X, const float* __restrict__ W,
                         const float* __restrict__ B, float* __restrict__ Y,
                         int M, int I, int O, int trans, int relu) {
  int i = blockIdx.x * blockDim.x + threadIdx.x;
  if (i >= M * O) return;
  int m = i / O, o = i - m * O;
  float s = B ? B[o] : 0.0f;
  const float* x = X + (size_t)m * I;
  if (trans) {
    const float* w = W + (size_t)o * I;
    for (int k = 0; k < I; ++k) s += x[k] * w[k];
  } else {
    for (int k = 0; k < I; ++k) s += x[k] * W[(size_t)k * O + o];
  }
  if (relu && s < 0.0f) s = 0.0f;
  Y[i] = s;
}

// comb[b,i,j,d] = E[b,i,d] * E[b,j,d]
__global__ void k_pairwise(const float* __restrict__ E, float* __restrict__ Y,
                           int B, int A, int D) {
  int idx = blockIdx.x * blockDim.x + threadIdx.x;
  int total = B * A * A * D;
  if (idx >= total) return;
  int d = idx % D; int tt = idx / D;
  int j = tt % A; tt /= A;
  int i2 = tt % A; int b = tt / A;
  Y[idx] = E[((size_t)b * A + i2) * D + d] * E[((size_t)b * A + j) * D + d];
}

// Y[b,j,d] = sum_i X[b,i,j,d]
__global__ void k_sumi(const float* __restrict__ X, float* __restrict__ Y,
                       int B, int A, int D) {
  int idx = blockIdx.x * blockDim.x + threadIdx.x;
  if (idx >= B * A * D) return;
  int d = idx % D; int tt = idx / D;
  int j = tt % A; int b = tt / A;
  float s = 0.0f;
  for (int i2 = 0; i2 < A; ++i2)
    s += X[(((size_t)b * A + i2) * A + j) * D + d];
  Y[idx] = s;
}

// Y[b,a,d] = U[b,a,d] * sum_{a'} V[b,a',d]
__global__ void k_aggmul(const float* __restrict__ U, const float* __restrict__ V,
                         float* __restrict__ Y, int B, int A, int D) {
  int idx = blockIdx.x * blockDim.x + threadIdx.x;
  if (idx >= B * A * D) return;
  int d = idx % D; int tt = idx / D; int b = tt / A;
  float s = 0.0f;
  for (int a2 = 0; a2 < A; ++a2) s += V[((size_t)b * A + a2) * D + d];
  Y[idx] = U[idx] * s;
}

// GRU gate combine (torch GRUCell math, gate order r,z,n)
__global__ void k_gru(const float* __restrict__ GI, const float* __restrict__ GH,
                      const float* __restrict__ H, float* __restrict__ HN,
                      int M, int D) {
  int i = blockIdx.x * blockDim.x + threadIdx.x;
  if (i >= M * D) return;
  int m = i / D, d = i - m * D;
  const float* gi = GI + (size_t)m * 3 * D;
  const float* gh = GH + (size_t)m * 3 * D;
  float r = 1.0f / (1.0f + expf(-(gi[d] + gh[d])));
  float z = 1.0f / (1.0f + expf(-(gi[D + d] + gh[D + d])));
  float nn = tanhf(gi[2 * D + d] + r * gh[2 * D + d]);
  HN[i] = (1.0f - z) * nn + z * H[i];
}

// P[b,k] = softmax(L[b,:])[k] * (1 + 0.5k)
__global__ void k_softmax_scale(const float* __restrict__ L, float* __restrict__ P,
                                int B, int K) {
  int b = blockIdx.x * blockDim.x + threadIdx.x;
  if (b >= B) return;
  const float* l = L + (size_t)b * K;
  float mx = l[0];
  for (int k = 1; k < K; ++k) mx = l[k] > mx ? l[k] : mx;
  float s = 0.0f;
  for (int k = 0; k < K; ++k) s += expf(l[k] - mx);
  for (int k = 0; k < K; ++k)
    P[(size_t)b * K + k] = expf(l[k] - mx) / s * (1.0f + 0.5f * (float)k);
}

// Y[b,d] = sum_a FI[b,a,d] + sum_a FU[b,a,d]
__global__ void k_interattr(const float* __restrict__ FI, const float* __restrict__ FU,
                            float* __restrict__ Y, int B, int A, int D) {
  int i = blockIdx.x * blockDim.x + threadIdx.x;
  if (i >= B * D) return;
  int b = i / D, d = i - b * D;
  float s = 0.0f;
  for (int a = 0; a < A; ++a)
    s += FI[((size_t)b * A + a) * D + d] + FU[((size_t)b * A + a) * D + d];
  Y[i] = s;
}

__global__ void k_add2(const float* __restrict__ X, const float* __restrict__ Y,
                       float* __restrict__ Z, int n) {
  int i = blockIdx.x * blockDim.x + threadIdx.x;
  if (i < n) Z[i] = X[i] + Y[i];
}

// ---------------------------------------------------------------------------
// Host orchestration
// ---------------------------------------------------------------------------
static inline int cdiv(int a, int b) { return (a + b - 1) / b; }
static inline size_t cdivz(size_t a, size_t b) { return (a + b - 1) / b; }

static void cvt16(hipStream_t s, const float* x, _Float16* y, size_t n) {
  k_cvt_f16<<<(unsigned)cdivz(n, 256), 256, 0, s>>>(x, y, n);
}

static void lin(hipStream_t s, const float* X, const float* W, const float* B,
                float* Y, int M, int I, int O, int trans, int relu) {
  k_linear<<<cdiv(M * O, 256), 256, 0, s>>>(X, W, B, Y, M, I, O, trans, relu);
}

// Convert weight to f16 and run the WMMA conv (fused BN/resid/ReLU)
static void conv(hipStream_t s, _Float16* wslot, const float* w_f32,
                 const _Float16* X, const float* bnScale, const float* bnBias,
                 const _Float16* resid, _Float16* Y,
                 int Nb, int Ci, int Hi, int Wi, int Co,
                 int R, int S, int stride, int pad, int relu,
                 int* HoOut, int* WoOut) {
  int Ho = (Hi + 2 * pad - R) / stride + 1;
  int Wo = (Wi + 2 * pad - S) / stride + 1;
  int K = Ci * R * S;
  cvt16(s, w_f32, wslot, (size_t)Co * K);
  int M = Nb * Ho * Wo;
  dim3 grid(cdiv(M, BM), Co / BNN);
  k_conv_wmma<<<grid, 256, 0, s>>>(X, wslot, bnScale, bnBias, resid, Y,
                                   Nb, Ci, Hi, Wi, Co, Ho, Wo, R, S, stride, pad, relu);
  if (HoOut) *HoOut = Ho;
  if (WoOut) *WoOut = Wo;
}

struct ConvBlk {
  const float *bn1b, *bn1s, *bn2b, *bn2s, *bn3b, *bn3s, *bndb, *bnds;
  const float *w1, *w2, *w3, *wd;
};
struct GruP { const float *bhh, *bih, *whh, *wih; };
struct MlpP { const float *b1, *b2, *w1, *w2; };

extern "C" void kernel_launch(void* const* d_in, const int* in_sizes, int n_in,
                              void* d_out, int out_size, void* d_ws, size_t ws_size,
                              hipStream_t stream) {
  (void)in_sizes; (void)n_in; (void)out_size; (void)ws_size;
  const int B = 32, A = 11, D = 64, H = 256, NB = 9, NPT = 50;

  // ---- JAX pytree flatten order: dict keys sorted, lists in sequence ----
  int c = 0;
  auto F = [&](int i) { return (const float*)d_in[i]; };
  const float* images = F(c++);                         // 'images'
  // 'params' subtree (alphabetical):
  const float* attr_corr_b = F(c++);                    // attr_corr.b
  const float* attr_corr_w = F(c++);                    // attr_corr.w
  // backbone.layers (list of 4 lists of block dicts)
  ConvBlk blocks[16];
  const int stageN[4] = {3, 4, 6, 3};
  int nbk = 0;
  for (int st = 0; st < 4; ++st) {
    for (int b = 0; b < stageN[st]; ++b) {
      ConvBlk bl = {};
      bl.bn1b = F(c++); bl.bn1s = F(c++);               // bn1.{bias,scale}
      bl.bn2b = F(c++); bl.bn2s = F(c++);
      bl.bn3b = F(c++); bl.bn3s = F(c++);
      if (b == 0) { bl.bndb = F(c++); bl.bnds = F(c++); }
      bl.w1 = F(c++); bl.w2 = F(c++); bl.w3 = F(c++);
      if (b == 0) bl.wd = F(c++);
      blocks[nbk++] = bl;
    }
  }
  const float* stem_bn_b = F(c++);                      // backbone.stem_bn.bias
  const float* stem_bn_s = F(c++);                      // backbone.stem_bn.scale
  const float* stem_w = F(c++);                         // backbone.stem_w
  const float* fc_aes_b = F(c++);  const float* fc_aes_w = F(c++);
  const float* fc_attr_b = F(c++); const float* fc_attr_w = F(c++);
  GruP gimg = { F(c), F(c+1), F(c+2), F(c+3) }; c += 4; // gru_img.{bhh,bih,whh,wih}
  GruP gusr = { F(c), F(c+1), F(c+2), F(c+3) }; c += 4;
  MlpP mint_img = { F(c), F(c+1), F(c+2), F(c+3) }; c += 4; // int_img.{b1,b2,w1,w2}
  MlpP mint_usr = { F(c), F(c+1), F(c+2), F(c+3) }; c += 4;
  MlpP mdist    = { F(c), F(c+1), F(c+2), F(c+3) }; c += 4; // mlp_dist
  MlpP mnode_i  = { F(c), F(c+1), F(c+2), F(c+3) }; c += 4; // node_attr_img
  MlpP mnode_u  = { F(c), F(c+1), F(c+2), F(c+3) }; c += 4; // node_attr_user
  const float* traits = F(c++);                         // 'personal_traits'

  // ---- workspace bump allocator ----
  char* wp = (char*)d_ws;
  auto alloc = [&](size_t bytes) -> void* {
    void* r = (void*)wp;
    wp += (bytes + 255) & ~(size_t)255;
    return r;
  };
  const size_t ACT = (size_t)60 << 20;  // per activation buffer (f16, max 49MB)
  _Float16* imgH  = (_Float16*)alloc((size_t)B * 3 * 224 * 224 * 2);
  _Float16* actX  = (_Float16*)alloc(ACT);
  _Float16* actA  = (_Float16*)alloc(ACT);
  _Float16* actB2 = (_Float16*)alloc(ACT);
  _Float16* actC  = (_Float16*)alloc(ACT);
  _Float16* actSC = (_Float16*)alloc(ACT);
  _Float16* wslot = (_Float16*)alloc((size_t)6 << 20);
  float* feat       = (float*)alloc((size_t)B * 2048 * 4);
  float* logit      = (float*)alloc((size_t)B * NB * 4);
  float* attrlog    = (float*)alloc((size_t)B * A * 4);
  float* t256       = (float*)alloc((size_t)B * A * A * H * 4); // biggest MLP hidden
  float* attr_img   = (float*)alloc((size_t)B * A * D * 4);
  float* attr_usr   = (float*)alloc((size_t)B * A * D * 4);
  float* comb       = (float*)alloc((size_t)B * A * A * D * 4);
  float* mo         = (float*)alloc((size_t)B * A * A * D * 4);
  float* int_img    = (float*)alloc((size_t)B * A * D * 4);
  float* int_usr    = (float*)alloc((size_t)B * A * D * 4);
  float* agg_img    = (float*)alloc((size_t)B * A * D * 4);
  float* agg_usr    = (float*)alloc((size_t)B * A * D * 4);
  float* gi         = (float*)alloc((size_t)B * A * 3 * D * 4);
  float* gh         = (float*)alloc((size_t)B * A * 3 * D * 4);
  float* hbuf       = (float*)alloc((size_t)B * A * D * 4);
  float* hbuf2      = (float*)alloc((size_t)B * A * D * 4);
  float* fused_img  = (float*)alloc((size_t)B * A * D * 4);
  float* fused_usr  = (float*)alloc((size_t)B * A * D * 4);
  float* inter      = (float*)alloc((size_t)B * D * 4);
  float* interv     = (float*)alloc((size_t)B * 4);
  float* pscaled    = (float*)alloc((size_t)B * NB * 4);
  float* direct     = (float*)alloc((size_t)B * 4);

  // ============================ ResNet-50 ============================
  cvt16(stream, images, imgH, (size_t)B * 3 * 224 * 224);
  int Hi, Wi;
  // stem: 7x7 s2 p3, Co=64, BN+ReLU fused
  conv(stream, wslot, stem_w, imgH, stem_bn_s, stem_bn_b, nullptr, actA,
       B, 3, 224, 224, 64, 7, 7, 2, 3, 1, &Hi, &Wi);              // 112x112
  // maxpool 3x3 s2 p1
  {
    int Ho = Hi / 2, Wo = Wi / 2;
    int total = B * 64 * Ho * Wo;
    k_maxpool3s2<<<cdiv(total, 256), 256, 0, stream>>>(actA, actX, B * 64, Hi, Wi, Ho, Wo);
    Hi = Ho; Wi = Wo;                                             // 56x56
  }
  int Cin = 64;
  const int mids[4] = {64, 128, 256, 512};
  nbk = 0;
  for (int st = 0; st < 4; ++st) {
    for (int b = 0; b < stageN[st]; ++b) {
      const ConvBlk& bl = blocks[nbk++];
      int mid = mids[st];
      int stride = (b == 0 && st > 0) ? 2 : 1;
      int Ho, Wo;
      const _Float16* sc;
      if (b == 0) {  // downsample: 1x1 stride, BN, no relu
        conv(stream, wslot, bl.wd, actX, bl.bnds, bl.bndb, nullptr, actSC,
             B, Cin, Hi, Wi, mid * 4, 1, 1, stride, 0, 0, &Ho, &Wo);
        sc = actSC;
      } else {
        sc = actX;
      }
      // conv1: 1x1 s1, BN+ReLU
      conv(stream, wslot, bl.w1, actX, bl.bn1s, bl.bn1b, nullptr, actA,
           B, Cin, Hi, Wi, mid, 1, 1, 1, 0, 1, nullptr, nullptr);
      // conv2: 3x3 stride, BN+ReLU
      conv(stream, wslot, bl.w2, actA, bl.bn2s, bl.bn2b, nullptr, actB2,
           B, mid, Hi, Wi, mid, 3, 3, stride, 1, 1, &Ho, &Wo);
      // conv3: 1x1 s1, BN + residual + ReLU
      conv(stream, wslot, bl.w3, actB2, bl.bn3s, bl.bn3b, sc, actC,
           B, mid, Ho, Wo, mid * 4, 1, 1, 1, 0, 1, nullptr, nullptr);
      // rotate: actC becomes new X
      _Float16* tmp = actX; actX = actC; actC = tmp;
      Cin = mid * 4; Hi = Ho; Wi = Wo;
    }
  }
  // global average pool -> feat [B, 2048] f32
  k_avgpool<<<cdiv(B * 2048, 256), 256, 0, stream>>>(actX, feat, B * 2048, Hi * Wi);

  // ============================== Head ==============================
  lin(stream, feat, fc_aes_w, fc_aes_b, logit, B, 2048, NB, 0, 0);
  lin(stream, feat, fc_attr_w, fc_attr_b, attrlog, B, 2048, A, 0, 0);

  // node attribute MLPs
  lin(stream, attrlog, mnode_i.w1, mnode_i.b1, t256, B, A, H, 0, 1);
  lin(stream, t256, mnode_i.w2, mnode_i.b2, attr_img, B, H, A * D, 0, 0);
  lin(stream, traits, mnode_u.w1, mnode_u.b1, t256, B, NPT, H, 0, 1);
  lin(stream, t256, mnode_u.w2, mnode_u.b2, attr_usr, B, H, A * D, 0, 0);

  // internal interaction (img)
  int combN = B * A * A * D;
  k_pairwise<<<cdiv(combN, 256), 256, 0, stream>>>(attr_img, comb, B, A, D);
  lin(stream, comb, mint_img.w1, mint_img.b1, t256, B * A * A, D, H, 0, 1);
  lin(stream, t256, mint_img.w2, mint_img.b2, mo, B * A * A, H, D, 0, 0);
  k_sumi<<<cdiv(B * A * D, 256), 256, 0, stream>>>(mo, int_img, B, A, D);
  // internal interaction (user)
  k_pairwise<<<cdiv(combN, 256), 256, 0, stream>>>(attr_usr, comb, B, A, D);
  lin(stream, comb, mint_usr.w1, mint_usr.b1, t256, B * A * A, D, H, 0, 1);
  lin(stream, t256, mint_usr.w2, mint_usr.b2, mo, B * A * A, H, D, 0, 0);
  k_sumi<<<cdiv(B * A * D, 256), 256, 0, stream>>>(mo, int_usr, B, A, D);

  // external interaction
  k_aggmul<<<cdiv(B * A * D, 256), 256, 0, stream>>>(attr_usr, attr_img, agg_usr, B, A, D);
  k_aggmul<<<cdiv(B * A * D, 256), 256, 0, stream>>>(attr_img, attr_usr, agg_img, B, A, D);

  // InterFusion: three chained GRU cells over {init, internal, external}
  const int M = B * A;
  auto interfusion = [&](const float* init, const float* internal, const float* external,
                         const GruP& g, float* out) {
    k_fill0<<<cdiv(M * D, 256), 256, 0, stream>>>(hbuf, M * D);
    const float* seq[3] = { init, internal, external };
    float* hc = hbuf; float* hn = hbuf2;
    for (int s2 = 0; s2 < 3; ++s2) {
      lin(stream, seq[s2], g.wih, g.bih, gi, M, D, 3 * D, 1, 0); // x @ wih.T + bih
      lin(stream, hc,      g.whh, g.bhh, gh, M, D, 3 * D, 1, 0); // h @ whh.T + bhh
      k_gru<<<cdiv(M * D, 256), 256, 0, stream>>>(gi, gh, hc, hn, M, D);
      float* t = hc; hc = hn; hn = t;
    }
    // copy hc -> out via add with zero-fill (avoid memcpy during capture)
    k_fill0<<<cdiv(M * D, 256), 256, 0, stream>>>(hn, M * D);
    k_add2<<<cdiv(M * D, 256), 256, 0, stream>>>(hc, hn, out, M * D);
  };
  interfusion(attr_img, int_img, agg_img, gimg, fused_img);
  interfusion(attr_usr, int_usr, agg_usr, gusr, fused_usr);

  // inter = (sum_a fused_img + sum_a fused_usr) @ attr_corr_w + b
  k_interattr<<<cdiv(B * D, 256), 256, 0, stream>>>(fused_img, fused_usr, inter, B, A, D);
  lin(stream, inter, attr_corr_w, attr_corr_b, interv, B, D, 1, 0, 0);

  // direct = MLP(softmax(logit) * scale)
  k_softmax_scale<<<cdiv(B, 64), 64, 0, stream>>>(logit, pscaled, B, NB);
  lin(stream, pscaled, mdist.w1, mdist.b1, t256, B, NB, H, 0, 1);
  lin(stream, t256, mdist.w2, mdist.b2, direct, B, H, 1, 0, 0);

  // out = inter + direct   -> d_out [B,1] f32
  k_add2<<<cdiv(B, 64), 64, 0, stream>>>(interv, direct, (float*)d_out, B);
}
